// EnhancedTarotInterpreter_51737176047636
// MI455X (gfx1250) — matrix operand order repaired and checked by reading, not compiled, not run on hardware
//
#include <hip/hip_runtime.h>
#include <hip/hip_bf16.h>
#include <math.h>

typedef __attribute__((ext_vector_type(16))) _Float16 v16h;
typedef __attribute__((ext_vector_type(8)))  _Float16 v8h;
typedef __attribute__((ext_vector_type(4)))  _Float16 v4h;
typedef __attribute__((ext_vector_type(8)))  float    v8f;

#define WMMA_F16(a,b,c) __builtin_amdgcn_wmma_f32_16x16x32_f16(false,(a),false,(b),(short)0,(c),false,false)

__device__ __forceinline__ v16h frag_cat(v8h lo, v8h hi) {
    return __builtin_shufflevector(lo, hi, 0,1,2,3,4,5,6,7,8,9,10,11,12,13,14,15);
}
// A-fragment (16x32 f16): lane&15 = M row; element idx -> K = 16g + idx
__device__ __forceinline__ v16h fragA(const _Float16* p) {     // p -> row base + 16g
    v8h lo = *(const v8h*)p;
    v8h hi = *(const v8h*)(p + 8);
    return frag_cat(lo, hi);
}
// B-fragment (32x16 f16): lane&15 = N col; element idx -> K = 16g + idx
__device__ __forceinline__ v16h fragB(const _Float16* p) {     // p -> row base + 16g
    v8h lo = *(const v8h*)p;
    v8h hi = *(const v8h*)(p + 8);
    return frag_cat(lo, hi);
}
__device__ __forceinline__ float halfmax(float v) {
    #pragma unroll
    for (int m = 1; m < 16; m <<= 1) v = fmaxf(v, __shfl_xor(v, m, 32));
    return v;
}
__device__ __forceinline__ float halfsum(float v) {
    #pragma unroll
    for (int m = 1; m < 16; m <<= 1) v += __shfl_xor(v, m, 32);
    return v;
}
__device__ __forceinline__ float wavesum(float v) {
    #pragma unroll
    for (int m = 1; m < 32; m <<= 1) v += __shfl_xor(v, m, 32);
    return v;
}
__device__ __forceinline__ float gelu_exact(float x) {
    return 0.5f * x * (1.0f + erff(x * 0.70710678118654752f));
}

// ---------------------------------------------------------------------------
// Tiled WMMA GEMM, f16 in / f32 accumulate:  C = act(A[M,K] * W[N,K]^T + bias)
// REQUIRES: M % 64 == 0, K % 32 == 0 (all call sites; only N ragged)
// block 256 (8 waves, 2x4), tile 64(M) x 128(N), BK=32. Pure-copy staging.
// ---------------------------------------------------------------------------
__global__ void gemm_wmma_f16(const _Float16* __restrict__ A, const _Float16* __restrict__ W,
                              const float* __restrict__ bias, float* __restrict__ Cf,
                              _Float16* __restrict__ Ch, int M, int N, int K, int act) {
    constexpr int LD = 40;  // halves; row stride 80B keeps b128 frag loads 16B aligned
    __shared__ __attribute__((aligned(16))) _Float16 As[64 * LD];
    __shared__ __attribute__((aligned(16))) _Float16 Bs[128 * LD];

    const int tid  = threadIdx.x;
    const int wave = tid >> 5, lane = tid & 31;
    const int g = lane >> 4, ln = lane & 15;
    const int wm = (wave >> 2) * 32;
    const int wn = (wave & 3) * 32;
    const int m0 = blockIdx.y * 64;
    const int n0 = blockIdx.x * 128;
    const int ra = tid >> 2;      // 0..63 staging row
    const int c8 = tid & 3;       // v8h column group (halves c8*8 .. c8*8+7)

    v8f acc[2][2];
    #pragma unroll
    for (int i = 0; i < 2; i++)
        #pragma unroll
        for (int j = 0; j < 2; j++)
            #pragma unroll
            for (int v = 0; v < 8; v++) acc[i][j][v] = 0.0f;

    for (int k0 = 0; k0 < K; k0 += 32) {
        // ---- pure f16 copies, loads issued before the barrier ----
        v8h av = *(const v8h*)(A + (size_t)(m0 + ra) * K + k0 + c8 * 8);
        int n1 = n0 + ra,      nc1 = n1 < N ? n1 : N - 1;   // clamp: in-bounds, no branch
        int n2 = n0 + ra + 64, nc2 = n2 < N ? n2 : N - 1;
        v8h bv0 = *(const v8h*)(W + (size_t)nc1 * K + k0 + c8 * 8);
        v8h bv1 = *(const v8h*)(W + (size_t)nc2 * K + k0 + c8 * 8);
        __syncthreads();                       // previous iteration's frag reads done
        *(v8h*)&As[ra * LD + c8 * 8]        = av;
        *(v8h*)&Bs[ra * LD + c8 * 8]        = bv0;
        *(v8h*)&Bs[(ra + 64) * LD + c8 * 8] = bv1;
        __syncthreads();

        v16h a0 = fragA(&As[(wm + ln) * LD + 16 * g]);
        v16h a1 = fragA(&As[(wm + 16 + ln) * LD + 16 * g]);
        v16h b0 = fragB(&Bs[(wn + ln) * LD + 16 * g]);
        v16h b1 = fragB(&Bs[(wn + 16 + ln) * LD + 16 * g]);

        acc[0][0] = WMMA_F16(a0, b0, acc[0][0]);
        acc[0][1] = WMMA_F16(a0, b1, acc[0][1]);
        acc[1][0] = WMMA_F16(a1, b0, acc[1][0]);
        acc[1][1] = WMMA_F16(a1, b1, acc[1][1]);
    }

    #pragma unroll
    for (int mt = 0; mt < 2; mt++)
        #pragma unroll
        for (int nt = 0; nt < 2; nt++) {
            int n = n0 + wn + nt * 16 + ln;
            if (n < N) {
                float bb = bias ? bias[n] : 0.0f;
                #pragma unroll
                for (int v = 0; v < 8; v++) {
                    int m = m0 + wm + mt * 16 + 8 * g + v;   // C layout: M = v + 8g
                    float f = acc[mt][nt][v] + bb;
                    if (act == 1) f = fmaxf(f, 0.0f);
                    if (Cf) Cf[(size_t)m * N + n] = f;
                    if (Ch) Ch[(size_t)m * N + n] = (_Float16)f;
                }
            }
        }
}

// ---------------------------------------------------------------------------
// Causal flash self-attention, all-f16 operands, fp32 softmax/accum.
// qkv: f16 [B,S,768] (Q|K|V), out: f16 [B,S,256]
// grid (S/128, H, B), block 256 = 8 waves; wave handles 16 query rows.
// ---------------------------------------------------------------------------
__global__ void attn_causal(const _Float16* __restrict__ qkv, _Float16* __restrict__ aout) {
    constexpr int LD = 40;
    __shared__ __attribute__((aligned(16))) _Float16 Ks[32 * LD];
    __shared__ __attribute__((aligned(16))) _Float16 Vt[32 * LD];      // transposed [hd][key]
    __shared__ __attribute__((aligned(16))) _Float16 Ps[8][16 * LD];   // per-wave P scratch

    const int tid = threadIdx.x;
    const int wave = tid >> 5, lane = tid & 31;
    const int g = lane >> 4, ln = lane & 15;
    const int blk = blockIdx.x, h = blockIdx.y, b = blockIdx.z;
    const int qrow0 = blk * 128 + wave * 16;
    const int r  = tid >> 3;      // 0..31
    const int c4 = tid & 7;       // halves c4*4..c4*4+3
    const size_t base = (size_t)b * 1024;
    const float scale = 0.17677669529663687f;   // 1/sqrt(32), applied to scores

    // Q A-fragment loaded DIRECTLY from global (f16 rows are frag-aligned)
    const _Float16* qp = qkv + (base + qrow0 + ln) * 768 + h * 32 + 16 * g;
    v16h qa = fragA(qp);

    v8f o0, o1, z;
    float mi[8], li[8];
    #pragma unroll
    for (int v = 0; v < 8; v++) { o0[v] = 0.0f; o1[v] = 0.0f; z[v] = 0.0f; mi[v] = -1e30f; li[v] = 0.0f; }

    const int kcmax = 4 * blk + 4;   // key chunks of 32 covering keys < 128*(blk+1)
    for (int kc = 0; kc < kcmax; kc++) {
        // stage K (row-major) + V (transposed): pure f16 copies, loads before barrier
        const _Float16* rp = qkv + (base + kc * 32 + r) * 768 + h * 32 + c4 * 4;
        v4h kh = *(const v4h*)(rp + 256);
        v4h vh = *(const v4h*)(rp + 512);
        __syncthreads();
        *(v4h*)&Ks[r * LD + c4 * 4] = kh;
        #pragma unroll
        for (int i = 0; i < 4; i++) Vt[(c4 * 4 + i) * LD + r] = vh[i];
        __syncthreads();

        if (kc * 32 <= qrow0 + 15) {
            v16h bk0 = fragB(&Ks[ln * LD + 16 * g]);
            v16h bk1 = fragB(&Ks[(16 + ln) * LD + 16 * g]);
            v8f s0 = WMMA_F16(qa, bk0, z);
            v8f s1 = WMMA_F16(qa, bk1, z);

            #pragma unroll
            for (int v = 0; v < 8; v++) {       // online softmax, row M = 8g+v
                int qidx = qrow0 + 8 * g + v;
                int ki = kc * 32 + ln;
                float x0 = (ki      > qidx) ? -1e30f : s0[v] * scale;
                float x1 = (ki + 16 > qidx) ? -1e30f : s1[v] * scale;
                float rm = halfmax(fmaxf(x0, x1));
                float mn = fmaxf(mi[v], rm);
                float al = __expf(mi[v] - mn);
                float p0 = __expf(x0 - mn), p1 = __expf(x1 - mn);
                float rs = halfsum(p0 + p1);
                li[v] = li[v] * al + rs;
                mi[v] = mn;
                o0[v] *= al; o1[v] *= al;       // C-frag rows align with score rows
                Ps[wave][(8 * g + v) * LD + ln]      = (_Float16)p0;
                Ps[wave][(8 * g + v) * LD + 16 + ln] = (_Float16)p1;
            }
            // same-wave DS ops are in-order: safe to reload P in A-layout
            v16h pa  = fragA(&Ps[wave][ln * LD + 16 * g]);
            v16h bv0 = fragB(&Vt[ln * LD + 16 * g]);
            v16h bv1 = fragB(&Vt[(16 + ln) * LD + 16 * g]);
            o0 = WMMA_F16(pa, bv0, o0);
            o1 = WMMA_F16(pa, bv1, o1);
        }
    }

    #pragma unroll
    for (int v = 0; v < 8; v++) {
        int qidx = qrow0 + 8 * g + v;
        float inv = 1.0f / li[v];
        size_t ob = (base + qidx) * 256 + h * 32;
        aout[ob + ln]      = (_Float16)(o0[v] * inv);
        aout[ob + 16 + ln] = (_Float16)(o1[v] * inv);
    }
}

// ---------------------------------------------------------------------------
// small helpers
// ---------------------------------------------------------------------------
__global__ void f32_to_f16(const float* __restrict__ x, _Float16* __restrict__ y, int n) {
    int i = blockIdx.x * 256 + threadIdx.x;
    if (i < n) y[i] = (_Float16)x[i];
}

__device__ __forceinline__ float block_sum_256(float v) {
    __shared__ float red[256];
    int t = threadIdx.x;
    red[t] = v; __syncthreads();
    for (int s = 128; s > 0; s >>= 1) { if (t < s) red[t] += red[t + s]; __syncthreads(); }
    float r = red[0]; __syncthreads();
    return r;
}

__global__ void embed_kernel(const int* __restrict__ tok, const float* __restrict__ temb,
                             const float* __restrict__ pemb, float* __restrict__ x,
                             _Float16* __restrict__ xh) {
    int i = blockIdx.x * 256 + threadIdx.x;
    int d = i & 255, bsi = i >> 8, s = bsi & 1023;
    float v = temb[(size_t)tok[bsi] * 256 + d] + pemb[(size_t)s * 256 + d];
    x[i] = v;
    xh[i] = (_Float16)v;
}

__global__ void enc_kernel(const float* __restrict__ t0, const float* __restrict__ t1,
                           const float* __restrict__ t2, const float* __restrict__ ew,
                           const float* __restrict__ eb, const float* __restrict__ lg,
                           const float* __restrict__ lb, float* __restrict__ cc) {
    int b = blockIdx.x, e = blockIdx.y, d = threadIdx.x;
    const float* xin = (e == 0 ? t0 : (e == 1 ? t1 : t2)) + b * 64;
    const float* w = ew + ((size_t)e * 256 + d) * 64;
    float t = eb[e * 256 + d];
    #pragma unroll 4
    for (int k = 0; k < 64; k++) t += xin[k] * w[k];
    float mean = block_sum_256(t) * (1.0f / 256.0f);
    float dv = t - mean;
    float var = block_sum_256(dv * dv) * (1.0f / 256.0f);
    float y = dv * rsqrtf(var + 1e-5f) * lg[e * 256 + d] + lb[e * 256 + d];
    cc[(size_t)b * 768 + e * 256 + d] = gelu_exact(y);
}

__global__ void fusion_kernel(const float* __restrict__ cc, const float* __restrict__ fw,
                              const float* __restrict__ fb, const float* __restrict__ lg,
                              const float* __restrict__ lb, float* __restrict__ memv) {
    int b = blockIdx.x, d = threadIdx.x;
    const float* xin = cc + (size_t)b * 768;
    const float* w = fw + (size_t)d * 768;
    float t = fb[d];
    #pragma unroll 4
    for (int k = 0; k < 768; k++) t += xin[k] * w[k];
    float mean = block_sum_256(t) * (1.0f / 256.0f);
    float dv = t - mean;
    float var = block_sum_256(dv * dv) * (1.0f / 256.0f);
    float y = dv * rsqrtf(var + 1e-5f) * lg[d] + lb[d];
    memv[(size_t)b * 256 + d] = gelu_exact(y);
}

// cross-attn with mem length 1: softmax==1 -> cvec[b] = (mem[b]@Wv.T+bv)@Wout.T+bout
__global__ void cvec_kernel(const float* __restrict__ memv, const float* __restrict__ win,
                            const float* __restrict__ bin, const float* __restrict__ wout,
                            const float* __restrict__ bout, float* __restrict__ cvec) {
    __shared__ float vbuf[256];
    int b = blockIdx.x, d = threadIdx.x;
    const float* xin = memv + (size_t)b * 256;
    const float* wv = win + (size_t)(512 + d) * 256;
    float t = bin[512 + d];
    #pragma unroll 4
    for (int k = 0; k < 256; k++) t += xin[k] * wv[k];
    vbuf[d] = t;
    __syncthreads();
    const float* wo = wout + (size_t)d * 256;
    float u = bout[d];
    #pragma unroll 4
    for (int k = 0; k < 256; k++) u += vbuf[k] * wo[k];
    cvec[(size_t)b * 256 + d] = u;
}

// in-place x[row] = LN(x[row] + r[sel]) ; writes fp32 master + f16 mirror
__global__ void resln_kernel(float* __restrict__ x, _Float16* __restrict__ xh,
                             const float* __restrict__ r, const float* __restrict__ gm,
                             const float* __restrict__ bt, int bcast) {
    int wave = threadIdx.x >> 5, lane = threadIdx.x & 31;
    int row = blockIdx.x * 8 + wave;
    const float* xr = x + (size_t)row * 256;
    const float* rr = r + (bcast ? (size_t)(row >> 10) * 256 : (size_t)row * 256);
    float v[8];
    #pragma unroll
    for (int j = 0; j < 8; j++) v[j] = xr[lane * 8 + j] + rr[lane * 8 + j];
    float s = 0.0f;
    #pragma unroll
    for (int j = 0; j < 8; j++) s += v[j];
    float mean = wavesum(s) * (1.0f / 256.0f);
    float q = 0.0f;
    #pragma unroll
    for (int j = 0; j < 8; j++) { float d = v[j] - mean; q += d * d; }
    float inv = rsqrtf(wavesum(q) * (1.0f / 256.0f) + 1e-5f);
    float* xw = x + (size_t)row * 256;
    _Float16* hw = xh + (size_t)row * 256;
    #pragma unroll
    for (int j = 0; j < 8; j++) {
        int c = lane * 8 + j;
        float y = (v[j] - mean) * inv * gm[c] + bt[c];
        xw[c] = y;
        hw[c] = (_Float16)y;
    }
}

// ---------------------------------------------------------------------------
extern "C" void kernel_launch(void* const* d_in, const int* in_sizes, int n_in,
                              void* d_out, int out_size, void* d_ws, size_t ws_size,
                              hipStream_t stream) {
    constexpr int B = 16, S = 1024, D = 256, V = 10000, NL = 3;
    constexpr int BS = B * S;

    const float* tract   = (const float*)d_in[0];
    const float* ctx     = (const float*)d_in[1];
    const float* card    = (const float*)d_in[2];
    const float* enc_w   = (const float*)d_in[3];
    const float* enc_b   = (const float*)d_in[4];
    const float* enc_lg  = (const float*)d_in[5];
    const float* enc_lb  = (const float*)d_in[6];
    const float* fus_w   = (const float*)d_in[7];
    const float* fus_b   = (const float*)d_in[8];
    const float* fus_lg  = (const float*)d_in[9];
    const float* fus_lb  = (const float*)d_in[10];
    const float* tok_emb = (const float*)d_in[11];
    const float* pos_emb = (const float*)d_in[12];
    const float* sa_in_w = (const float*)d_in[13];
    const float* sa_in_b = (const float*)d_in[14];
    const float* sa_o_w  = (const float*)d_in[15];
    const float* sa_o_b  = (const float*)d_in[16];
    const float* ca_in_w = (const float*)d_in[17];
    const float* ca_in_b = (const float*)d_in[18];
    const float* ca_o_w  = (const float*)d_in[19];
    const float* ca_o_b  = (const float*)d_in[20];
    const float* ln1_g   = (const float*)d_in[21];
    const float* ln1_b   = (const float*)d_in[22];
    const float* ln2_g   = (const float*)d_in[23];
    const float* ln2_b   = (const float*)d_in[24];
    const float* ln3_g   = (const float*)d_in[25];
    const float* ln3_b   = (const float*)d_in[26];
    const float* ffn_w1  = (const float*)d_in[27];
    const float* ffn_b1  = (const float*)d_in[28];
    const float* ffn_w2  = (const float*)d_in[29];
    const float* ffn_b2  = (const float*)d_in[30];
    const float* out_w   = (const float*)d_in[31];
    const float* out_b   = (const float*)d_in[32];
    const int*   prev    = (const int*)d_in[33];
    float* out = (float*)d_out;

    // fp32 workspace section
    float* ws = (float*)d_ws;
    size_t off = 0;
    float* X    = ws + off; off += (size_t)BS * D;     // fp32 residual master
    float* T0   = ws + off; off += (size_t)BS * D;     // proj / ffn2 temp
    float* CC   = ws + off; off += (size_t)B * 768;
    float* MEMV = ws + off; off += (size_t)B * 256;
    float* CVEC = ws + off; off += (size_t)B * 256;
    // f16 workspace section (16B aligned: all offsets multiples of 8 halves)
    _Float16* hb = (_Float16*)(ws + off);
    size_t ho = 0;
    _Float16* Xh    = hb + ho; ho += (size_t)BS * D;
    _Float16* QKVh  = hb + ho; ho += (size_t)BS * 3 * D;
    _Float16* AOUTh = hb + ho; ho += (size_t)BS * D;
    _Float16* HBh   = hb + ho; ho += (size_t)BS * 4 * D;
    _Float16* WsaIn = hb + ho; ho += (size_t)NL * 768 * 256;
    _Float16* WsaO  = hb + ho; ho += (size_t)NL * 256 * 256;
    _Float16* Wff1  = hb + ho; ho += (size_t)NL * 1024 * 256;
    _Float16* Wff2  = hb + ho; ho += (size_t)NL * 256 * 1024;
    _Float16* Wout  = hb + ho; ho += (size_t)V * 256;

    const dim3 blk256(256);
    // one-time (per launch) weight downcasts — ~10 MB total, negligible at 23.3 TB/s
    f32_to_f16<<<(NL * 768 * 256 + 255) / 256, blk256, 0, stream>>>(sa_in_w, WsaIn, NL * 768 * 256);
    f32_to_f16<<<(NL * 256 * 256 + 255) / 256, blk256, 0, stream>>>(sa_o_w, WsaO, NL * 256 * 256);
    f32_to_f16<<<(NL * 1024 * 256 + 255) / 256, blk256, 0, stream>>>(ffn_w1, Wff1, NL * 1024 * 256);
    f32_to_f16<<<(NL * 256 * 1024 + 255) / 256, blk256, 0, stream>>>(ffn_w2, Wff2, NL * 256 * 1024);
    f32_to_f16<<<(V * 256 + 255) / 256, blk256, 0, stream>>>(out_w, Wout, V * 256);

    embed_kernel<<<BS * D / 256, blk256, 0, stream>>>(prev, tok_emb, pos_emb, X, Xh);
    enc_kernel<<<dim3(B, 3), blk256, 0, stream>>>(tract, ctx, card, enc_w, enc_b, enc_lg, enc_lb, CC);
    fusion_kernel<<<B, blk256, 0, stream>>>(CC, fus_w, fus_b, fus_lg, fus_lb, MEMV);

    for (int l = 0; l < NL; l++) {
        // self-attention
        gemm_wmma_f16<<<dim3(6, 256), blk256, 0, stream>>>(
            Xh, WsaIn + (size_t)l * 768 * 256, sa_in_b + (size_t)l * 768,
            nullptr, QKVh, BS, 768, 256, 0);
        attn_causal<<<dim3(8, 8, 16), blk256, 0, stream>>>(QKVh, AOUTh);
        gemm_wmma_f16<<<dim3(2, 256), blk256, 0, stream>>>(
            AOUTh, WsaO + (size_t)l * 256 * 256, sa_o_b + (size_t)l * 256,
            T0, nullptr, BS, 256, 256, 0);
        resln_kernel<<<BS / 8, blk256, 0, stream>>>(X, Xh, T0, ln1_g + l * 256, ln1_b + l * 256, 0);
        // cross-attention over length-1 memory (softmax == 1)
        cvec_kernel<<<B, blk256, 0, stream>>>(
            MEMV, ca_in_w + (size_t)l * 768 * 256, ca_in_b + (size_t)l * 768,
            ca_o_w + (size_t)l * 256 * 256, ca_o_b + (size_t)l * 256, CVEC);
        resln_kernel<<<BS / 8, blk256, 0, stream>>>(X, Xh, CVEC, ln2_g + l * 256, ln2_b + l * 256, 1);
        // FFN (ReLU)
        gemm_wmma_f16<<<dim3(8, 256), blk256, 0, stream>>>(
            Xh, Wff1 + (size_t)l * 1024 * 256, ffn_b1 + (size_t)l * 1024,
            nullptr, HBh, BS, 1024, 256, 1);
        gemm_wmma_f16<<<dim3(2, 256), blk256, 0, stream>>>(
            HBh, Wff2 + (size_t)l * 256 * 1024, ffn_b2 + (size_t)l * 256,
            T0, nullptr, BS, 256, 1024, 0);
        resln_kernel<<<BS / 8, blk256, 0, stream>>>(X, Xh, T0, ln3_g + l * 256, ln3_b + l * 256, 0);
    }
    // final vocab projection (dominant GEMM: 16384 x 10000 x 256), fp32 out
    gemm_wmma_f16<<<dim3((V + 127) / 128, BS / 64), blk256, 0, stream>>>(
        Xh, Wout, out_b, out, nullptr, BS, V, 256, 0);

    (void)in_sizes; (void)n_in; (void)out_size; (void)ws_size;
}
